// MiniGridSeq2SeqTransformerVQ_73735998538506
// MI455X (gfx1250) — compile-verified, matrix operand order checked by reading, not run on hardware
//
#include <hip/hip_runtime.h>
#include <stdint.h>

// ---------------------------------------------------------------------------
// MI455X (gfx1250) implementation: bf16 WMMA GEMM engine + VALU glue kernels.
// Async global->LDS double-buffered pipeline + v_wmma_f32_16x16x32_bf16.
// ---------------------------------------------------------------------------

typedef __bf16 bf16_t;
typedef __attribute__((ext_vector_type(16))) __bf16 v16bf;
typedef __attribute__((ext_vector_type(8)))  __bf16 v8bf;
typedef __attribute__((ext_vector_type(8)))  float  v8f;
typedef int v4i_gcc __attribute__((vector_size(16)));

static constexpr int CB   = 128;   // batch
static constexpr int CT   = 20;    // seq len
static constexpr int CS   = 512;   // state dim
static constexpr int CD   = 1024;  // model dim
static constexpr int CH   = 16;    // heads
static constexpr int CL   = 4;     // layers
static constexpr int CK   = 2048;  // codebook entries
static constexpr int CDFF = 4096;  // ffn dim

// ---- gfx1250 async global->LDS path (guarded; fallback = load + ds_write) ----
#if defined(__has_builtin)
#  if __has_builtin(__builtin_amdgcn_global_load_async_to_lds_b128)
#    define USE_ASYNC_LDS 1
#  endif
#endif

static __device__ __forceinline__ void g2l_b128(const void* g, void* l) {
#ifdef USE_ASYNC_LDS
  __builtin_amdgcn_global_load_async_to_lds_b128(
      (__attribute__((address_space(1))) v4i_gcc*)(g),
      (__attribute__((address_space(3))) v4i_gcc*)(l),
      0, 0);
#else
  *(uint4*)l = *(const uint4*)g;
#endif
}

static __device__ __forceinline__ void g2l_wait0() {
#ifdef USE_ASYNC_LDS
#  if __has_builtin(__builtin_amdgcn_s_wait_asynccnt)
  __builtin_amdgcn_s_wait_asynccnt(0);
#  else
  asm volatile("s_wait_asynccnt 0" ::: "memory");
#  endif
#endif
}

static __device__ __forceinline__ void g2l_wait8() {
#ifdef USE_ASYNC_LDS
#  if __has_builtin(__builtin_amdgcn_s_wait_asynccnt)
  __builtin_amdgcn_s_wait_asynccnt(8);
#  else
  asm volatile("s_wait_asynccnt 8" ::: "memory");
#  endif
#endif
}

// ---------------------------------------------------------------------------
// WMMA GEMM: C[M,N] = A[M,K] * W[N,K]^T (+bias) (optional relu, bf16 mirror)
// A, W bf16 row-major (K contiguous).  M%128==0, N%128==0, K%64==0.
// Block: 256 threads (8 wave32s), tile 128x128, BK=64, double-buffered 64KB LDS.
// Wave grid 2(M)x4(N): each wave owns a 64x32 patch = 4x2 WMMA tiles.
// Pipeline: async-copy tile i+1 while computing tile i; s_wait_asynccnt<=8
// releases exactly the 8 outstanding copies of the current tile.
// ---------------------------------------------------------------------------
template <bool HAS_BIAS, bool RELU, bool HAS_CB>
__global__ void __launch_bounds__(256)
wmma_gemm_bf16(const bf16_t* __restrict__ A, const bf16_t* __restrict__ W,
               const float* __restrict__ bias, float* __restrict__ C,
               bf16_t* __restrict__ Cb, int M, int N, int K)
{
  __shared__ bf16_t As[2][128][64];
  __shared__ bf16_t Bs[2][128][64];
  const int tid  = threadIdx.x;
  const int lane = tid & 31;
  const int wave = tid >> 5;
  const int wm   = wave & 1;    // 0..1 -> 64-row slice of the 128-row tile
  const int wn   = wave >> 1;   // 0..3 -> 32-col slice of the 128-col tile
  const int m0   = blockIdx.y * 128;
  const int n0   = blockIdx.x * 128;

  v8f acc[4][2];
#pragma unroll
  for (int i = 0; i < 4; ++i)
#pragma unroll
    for (int j = 0; j < 2; ++j)
#pragma unroll
      for (int e = 0; e < 8; ++e) acc[i][j][e] = 0.0f;

  const int ntiles = K >> 6;

  auto issue = [&](int tile, int buf) {
    const int kbase = tile * 64;
    // A tile: 128 rows * 128B = 1024 x 16B segments -> 4 per thread
#pragma unroll
    for (int i = 0; i < 4; ++i) {
      int seg = tid + i * 256;
      int r = seg >> 3, c = (seg & 7) * 8;
      g2l_b128(A + (size_t)(m0 + r) * K + kbase + c, &As[buf][r][c]);
    }
    // B tile: 128 rows -> 1024 segments -> 4 per thread
#pragma unroll
    for (int i = 0; i < 4; ++i) {
      int seg = tid + i * 256;
      int r = seg >> 3, c = (seg & 7) * 8;
      g2l_b128(W + (size_t)(n0 + r) * K + kbase + c, &Bs[buf][r][c]);
    }
  };

  issue(0, 0);
  for (int it = 0; it < ntiles; ++it) {
    const int buf = it & 1;
    const bool more = (it + 1) < ntiles;
    if (more) issue(it + 1, (it + 1) & 1);   // overlap copy with compute
    if (more) g2l_wait8(); else g2l_wait0(); // current tile's 8 copies done
    __syncthreads();

#pragma unroll
    for (int kk = 0; kk < 2; ++kk) {
      // 16-bit A fragment layout: lanes 0-15 hold K {0..7,16..23},
      // lanes 16-31 hold K {8..15,24..31} (doc 7.12.2); B mirrored on N.
      const int kb = kk * 32 + ((lane & 16) ? 8 : 0);
      v16bf af[4], bfv[2];
#pragma unroll
      for (int mi = 0; mi < 4; ++mi) {
        const int r = wm * 64 + mi * 16 + (lane & 15);
        const v8bf* p0 = (const v8bf*)&As[buf][r][kb];
        const v8bf* p1 = (const v8bf*)&As[buf][r][kb + 16];
#pragma unroll
        for (int e = 0; e < 8; ++e) { af[mi][e] = (*p0)[e]; af[mi][e + 8] = (*p1)[e]; }
      }
#pragma unroll
      for (int ni = 0; ni < 2; ++ni) {
        const int r = wn * 32 + ni * 16 + (lane & 15);
        const v8bf* p0 = (const v8bf*)&Bs[buf][r][kb];
        const v8bf* p1 = (const v8bf*)&Bs[buf][r][kb + 16];
#pragma unroll
        for (int e = 0; e < 8; ++e) { bfv[ni][e] = (*p0)[e]; bfv[ni][e + 8] = (*p1)[e]; }
      }
#pragma unroll
      for (int mi = 0; mi < 4; ++mi)
#pragma unroll
        for (int ni = 0; ni < 2; ++ni)
          acc[mi][ni] = __builtin_amdgcn_wmma_f32_16x16x32_bf16(
              false, af[mi], false, bfv[ni], (short)0, acc[mi][ni], false, false);
    }
    __syncthreads();   // reads done before next issue overwrites this buffer
  }

  // C layout (16x16 f32): lanes 0-15 -> N=lane, M=reg; lanes 16-31 -> M=reg+8.
#pragma unroll
  for (int mi = 0; mi < 4; ++mi)
#pragma unroll
    for (int ni = 0; ni < 2; ++ni) {
      const int col = n0 + wn * 32 + ni * 16 + (lane & 15);
      float bv = 0.0f;
      if (HAS_BIAS) bv = bias[col];
      const int rb = m0 + wm * 64 + mi * 16 + ((lane & 16) ? 8 : 0);
#pragma unroll
      for (int r = 0; r < 8; ++r) {
        float v = acc[mi][ni][r] + bv;
        if (RELU) v = fmaxf(v, 0.0f);
        size_t o = (size_t)(rb + r) * N + col;
        C[o] = v;
        if (HAS_CB) Cb[o] = (bf16_t)v;
      }
    }
}

// ---------------------------------------------------------------------------
// fp32 -> bf16 conversion
// ---------------------------------------------------------------------------
__global__ void __launch_bounds__(256)
cvt_bf16_kernel(const float* __restrict__ s, bf16_t* __restrict__ d, long n)
{
  long i = (long)blockIdx.x * 256 + threadIdx.x;
  if (i < n) d[i] = (bf16_t)s[i];
}

// ---------------------------------------------------------------------------
// Tiny-T multihead attention (T<=20, hd=64). One block per (b,h,tq), 64 thr.
// ---------------------------------------------------------------------------
__global__ void __launch_bounds__(64)
attn_kernel(const float* __restrict__ q, int qs,
            const float* __restrict__ k, int ks,
            const float* __restrict__ v, int vs,
            float* __restrict__ out, int Tq, int Tk, int Hn, int causal)
{
  __shared__ float qsh[64];
  __shared__ float sc[64];
  const int bid = blockIdx.x;
  const int tq = bid % Tq;
  const int h  = (bid / Tq) % Hn;
  const int b  = bid / (Tq * Hn);
  const int d  = threadIdx.x;
  const float scale = 0.125f; // 1/sqrt(64)

  qsh[d] = q[(size_t)(b * Tq + tq) * qs + h * 64 + d];
  __syncthreads();

  float s = -1e30f;
  if (d < Tk && (!causal || d <= tq)) {
    const float* kr = k + (size_t)(b * Tk + d) * ks + h * 64;
    float a = 0.0f;
#pragma unroll 8
    for (int e = 0; e < 64; ++e) a += qsh[e] * kr[e];
    s = a * scale;
  }
  sc[d] = s;
  __syncthreads();

  float mx = -1e30f;
  for (int t = 0; t < Tk; ++t) mx = fmaxf(mx, sc[t]);
  float den = 0.0f;
  for (int t = 0; t < Tk; ++t) den += __expf(sc[t] - mx);
  float o = 0.0f;
  for (int t = 0; t < Tk; ++t) {
    float a = __expf(sc[t] - mx) / den;
    o += a * v[(size_t)(b * Tk + t) * vs + h * 64 + d];
  }
  out[(size_t)(b * Tq + tq) * (Hn * 64) + h * 64 + d] = o;
}

// ---------------------------------------------------------------------------
// LayerNorm(res + add) with affine; writes f32 + optional bf16 mirror.
// One block (256 thr) per row. Safe in-place (of may alias res).
// ---------------------------------------------------------------------------
__global__ void __launch_bounds__(256)
ln_kernel(const float* __restrict__ res, const float* __restrict__ add,
          const float* __restrict__ g, const float* __restrict__ b,
          float* __restrict__ of, bf16_t* __restrict__ ob, int Dn)
{
  __shared__ float red[256];
  const int row = blockIdx.x, tid = threadIdx.x;
  const float* rr = res + (size_t)row * Dn;
  const float* ar = add + (size_t)row * Dn;

  float s = 0.0f;
  for (int i = tid; i < Dn; i += 256) s += rr[i] + ar[i];
  red[tid] = s; __syncthreads();
  for (int st = 128; st > 0; st >>= 1) { if (tid < st) red[tid] += red[tid + st]; __syncthreads(); }
  const float mean = red[0] / Dn;
  __syncthreads();

  float s2 = 0.0f;
  for (int i = tid; i < Dn; i += 256) { float x = rr[i] + ar[i] - mean; s2 += x * x; }
  red[tid] = s2; __syncthreads();
  for (int st = 128; st > 0; st >>= 1) { if (tid < st) red[tid] += red[tid + st]; __syncthreads(); }
  const float rstd = rsqrtf(red[0] / Dn + 1e-5f);

  for (int i = tid; i < Dn; i += 256) {
    float x = (rr[i] + ar[i] - mean) * rstd * g[i] + b[i];
    of[(size_t)row * Dn + i] = x;
    if (ob) ob[(size_t)row * Dn + i] = (bf16_t)x;
  }
}

// ---------------------------------------------------------------------------
// Embedding add: x = se + act_emb[actions] + pos_emb
// ---------------------------------------------------------------------------
__global__ void __launch_bounds__(256)
embed_kernel(const float* __restrict__ se, const int* __restrict__ act,
             const float* __restrict__ aemb, const float* __restrict__ pos,
             float* __restrict__ xf, bf16_t* __restrict__ xb)
{
  int i = blockIdx.x * 256 + threadIdx.x;
  if (i >= CB * CT * CD) return;
  int d = i % CD;
  int t = (i / CD) % CT;
  int bt = i / CD;
  float v = se[i] + aemb[(size_t)act[bt] * CD + d] + pos[t * CD + d];
  xf[i] = v; xb[i] = (bf16_t)v;
}

// ---------------------------------------------------------------------------
// VQ helpers
// ---------------------------------------------------------------------------
__global__ void __launch_bounds__(256)
rownorm_kernel(const float* __restrict__ cb, float* __restrict__ cn, int Dn)
{
  __shared__ float red[256];
  const int row = blockIdx.x, tid = threadIdx.x;
  float s = 0.0f;
  for (int i = tid; i < Dn; i += 256) { float c = cb[(size_t)row * Dn + i]; s += c * c; }
  red[tid] = s; __syncthreads();
  for (int st = 128; st > 0; st >>= 1) { if (tid < st) red[tid] += red[tid + st]; __syncthreads(); }
  if (tid == 0) cn[row] = red[0];
}

__global__ void __launch_bounds__(256)
vq_argmin_kernel(const float* __restrict__ dots, const float* __restrict__ cn,
                 int* __restrict__ idx, int Kn)
{
  __shared__ float bv[256];
  __shared__ int   bi[256];
  const int row = blockIdx.x, tid = threadIdx.x;
  float best = 3.4e38f; int bidx = 0;
  for (int j = tid; j < Kn; j += 256) {
    float d = cn[j] - 2.0f * dots[(size_t)row * Kn + j];
    if (d < best) { best = d; bidx = j; }
  }
  bv[tid] = best; bi[tid] = bidx; __syncthreads();
  for (int st = 128; st > 0; st >>= 1) {
    if (tid < st) {
      if (bv[tid + st] < bv[tid] ||
          (bv[tid + st] == bv[tid] && bi[tid + st] < bi[tid])) {
        bv[tid] = bv[tid + st]; bi[tid] = bi[tid + st];
      }
    }
    __syncthreads();
  }
  if (tid == 0) idx[row] = bi[0];
}

__global__ void __launch_bounds__(256)
vq_quant_kernel(const float* __restrict__ x, const float* __restrict__ cb,
                const int* __restrict__ idx, float* __restrict__ qf,
                float* __restrict__ rowloss, int Dn)
{
  __shared__ float red[256];
  const int row = blockIdx.x, tid = threadIdx.x;
  const int j = idx[row];
  float s = 0.0f;
  for (int d = tid; d < Dn; d += 256) {
    float q = cb[(size_t)j * Dn + d];
    qf[(size_t)row * Dn + d] = q;
    float df = x[(size_t)row * Dn + d] - q;
    s += df * df;
  }
  red[tid] = s; __syncthreads();
  for (int st = 128; st > 0; st >>= 1) { if (tid < st) red[tid] += red[tid + st]; __syncthreads(); }
  if (tid == 0) rowloss[row] = red[0];
}

__global__ void __launch_bounds__(256)
reduce_kernel(const float* __restrict__ rl, int n, float scale, float* __restrict__ o)
{
  __shared__ float red[256];
  const int tid = threadIdx.x;
  float s = 0.0f;
  for (int i = tid; i < n; i += 256) s += rl[i];
  red[tid] = s; __syncthreads();
  for (int st = 128; st > 0; st >>= 1) { if (tid < st) red[tid] += red[tid + st]; __syncthreads(); }
  if (tid == 0) o[0] = red[0] * scale;
}

__global__ void __launch_bounds__(256)
idx_out_kernel(const int* __restrict__ idx, float* __restrict__ o, int n)
{
  int i = blockIdx.x * 256 + threadIdx.x;
  if (i < n) o[i] = (float)idx[i];
}

// ---------------------------------------------------------------------------
// Decoder bookkeeping
// ---------------------------------------------------------------------------
__global__ void __launch_bounds__(256)
set_col_seq_kernel(const float* __restrict__ seq, float* __restrict__ dec_in,
                   int tsrc, int tdst)
{
  int i = blockIdx.x * 256 + threadIdx.x;
  if (i >= CB * CD) return;
  int b = i / CD, d = i % CD;
  dec_in[((size_t)b * CT + tdst) * CD + d] = seq[((size_t)b * CT + tsrc) * CD + d];
}

__global__ void __launch_bounds__(256)
set_col_kernel(const float* __restrict__ src, float* __restrict__ dec_in, int t)
{
  int i = blockIdx.x * 256 + threadIdx.x;
  if (i >= CB * CD) return;
  int b = i / CD, d = i % CD;
  dec_in[((size_t)b * CT + t) * CD + d] = src[i];
}

__global__ void __launch_bounds__(256)
build_y_kernel(const float* __restrict__ dec_in, const float* __restrict__ pos,
               float* __restrict__ yf, bf16_t* __restrict__ yb, int Tc)
{
  int i = blockIdx.x * 256 + threadIdx.x;
  if (i >= CB * Tc * CD) return;
  int d = i % CD;
  int tau = (i / CD) % Tc;
  int b = i / (CD * Tc);
  float v = dec_in[((size_t)b * CT + tau) * CD + d] + pos[tau * CD + d];
  yf[i] = v; yb[i] = (bf16_t)v;
}

__global__ void __launch_bounds__(256)
build_mem_kernel(const float* __restrict__ quant, bf16_t* __restrict__ memb, int Tc)
{
  int i = blockIdx.x * 256 + threadIdx.x;
  if (i >= CB * Tc * CD) return;
  int d = i % CD;
  int tau = (i / CD) % Tc;
  int b = i / (CD * Tc);
  memb[i] = (bf16_t)quant[((size_t)b * CT + tau) * CD + d];
}

__global__ void __launch_bounds__(256)
gather_last_kernel(const float* __restrict__ y, bf16_t* __restrict__ lastb, int Tc)
{
  int i = blockIdx.x * 256 + threadIdx.x;
  if (i >= CB * CD) return;
  int b = i / CD, d = i % CD;
  lastb[i] = (bf16_t)y[((size_t)b * Tc + (Tc - 1)) * CD + d];
}

__global__ void __launch_bounds__(256)
store_pred_kernel(const float* __restrict__ pred, float* __restrict__ outp,
                  bf16_t* __restrict__ predb, int t)
{
  int i = blockIdx.x * 256 + threadIdx.x;
  if (i >= CB * CS) return;
  int b = i / CS, s = i % CS;
  float v = pred[i];
  outp[((size_t)b * (CT - 1) + t) * CS + s] = v;
  predb[i] = (bf16_t)v;
}

// ---------------------------------------------------------------------------
// Host orchestration
// ---------------------------------------------------------------------------
extern "C" void kernel_launch(void* const* d_in, const int* in_sizes, int n_in,
                              void* d_out, int out_size, void* d_ws, size_t ws_size,
                              hipStream_t stream)
{
  (void)in_sizes; (void)n_in; (void)out_size; (void)ws_size;

  const float* states    = (const float*)d_in[0];
  const int*   actions   = (const int*)  d_in[1];
  const float* enc_W     = (const float*)d_in[2];
  const float* enc_b     = (const float*)d_in[3];
  const float* act_emb   = (const float*)d_in[4];
  const float* pos_emb   = (const float*)d_in[5];
  const float* enc_qkv_w = (const float*)d_in[6];
  const float* enc_qkv_b = (const float*)d_in[7];
  const float* enc_out_w = (const float*)d_in[8];
  const float* enc_out_b = (const float*)d_in[9];
  const float* enc_ln1_g = (const float*)d_in[10];
  const float* enc_ln1_b = (const float*)d_in[11];
  const float* enc_ln2_g = (const float*)d_in[12];
  const float* enc_ln2_b = (const float*)d_in[13];
  const float* enc_ff1_w = (const float*)d_in[14];
  const float* enc_ff1_b = (const float*)d_in[15];
  const float* enc_ff2_w = (const float*)d_in[16];
  const float* enc_ff2_b = (const float*)d_in[17];
  const float* dsa_qkv_w = (const float*)d_in[18];
  const float* dsa_qkv_b = (const float*)d_in[19];
  const float* dsa_out_w = (const float*)d_in[20];
  const float* dsa_out_b = (const float*)d_in[21];
  const float* dca_qkv_w = (const float*)d_in[22];
  const float* dca_qkv_b = (const float*)d_in[23];
  const float* dca_out_w = (const float*)d_in[24];
  const float* dca_out_b = (const float*)d_in[25];
  const float* dln1_g    = (const float*)d_in[26];
  const float* dln1_b    = (const float*)d_in[27];
  const float* dln2_g    = (const float*)d_in[28];
  const float* dln2_b    = (const float*)d_in[29];
  const float* dln3_g    = (const float*)d_in[30];
  const float* dln3_b    = (const float*)d_in[31];
  const float* dff1_w    = (const float*)d_in[32];
  const float* dff1_b    = (const float*)d_in[33];
  const float* dff2_w    = (const float*)d_in[34];
  const float* dff2_b    = (const float*)d_in[35];
  const float* codebook  = (const float*)d_in[36];
  const float* rec_W     = (const float*)d_in[37];
  const float* rec_b     = (const float*)d_in[38];
  float* out = (float*)d_out;

  // ---- workspace carve ----
  char* wsb = (char*)d_ws;
  size_t cur = 0;
  auto alloc  = [&](size_t bytes) -> void* {
    void* p = wsb + cur; cur = (cur + bytes + 255) & ~(size_t)255; return p;
  };
  auto allocF = [&](size_t n) -> float*  { return (float*)alloc(n * sizeof(float)); };
  auto allocH = [&](size_t n) -> bf16_t* { return (bf16_t*)alloc(n * sizeof(bf16_t)); };

  auto cvt = [&](const float* s, bf16_t* d, size_t n) {
    cvt_bf16_kernel<<<(unsigned)((n + 255) / 256), 256, 0, stream>>>(s, d, (long)n);
  };
  auto gemm = [&](const bf16_t* A, const bf16_t* Wt, const float* bias,
                  float* C, bf16_t* Cb, int M, int N, int Kd, int relu) {
    dim3 g(N / 128, M / 128);
    if (bias) {
      if (relu) {
        if (Cb) wmma_gemm_bf16<true,  true,  true ><<<g, 256, 0, stream>>>(A, Wt, bias, C, Cb, M, N, Kd);
        else    wmma_gemm_bf16<true,  true,  false><<<g, 256, 0, stream>>>(A, Wt, bias, C, Cb, M, N, Kd);
      } else {
        if (Cb) wmma_gemm_bf16<true,  false, true ><<<g, 256, 0, stream>>>(A, Wt, bias, C, Cb, M, N, Kd);
        else    wmma_gemm_bf16<true,  false, false><<<g, 256, 0, stream>>>(A, Wt, bias, C, Cb, M, N, Kd);
      }
    } else {
      if (Cb) wmma_gemm_bf16<false, false, true ><<<g, 256, 0, stream>>>(A, Wt, bias, C, Cb, M, N, Kd);
      else    wmma_gemm_bf16<false, false, false><<<g, 256, 0, stream>>>(A, Wt, bias, C, Cb, M, N, Kd);
    }
  };
  auto ln = [&](const float* res, const float* add, const float* g, const float* b,
                float* of, bf16_t* ob, int M) {
    ln_kernel<<<M, 256, 0, stream>>>(res, add, g, b, of, ob, CD);
  };
  auto attn = [&](const float* q, int qs, const float* k, int ks,
                  const float* v, int vs, float* o, int Tq, int Tk, int causal) {
    attn_kernel<<<CB * CH * Tq, 64, 0, stream>>>(q, qs, k, ks, v, vs, o, Tq, Tk, CH, causal);
  };

  // ---- bf16 weight copies (decoder stack ~130MB bf16 -> L2 resident) ----
  bf16_t* wb_encW   = allocH((size_t)CD * CS);
  bf16_t* wb_eqkv   = allocH((size_t)CL * 3 * CD * CD);
  bf16_t* wb_eout   = allocH((size_t)CL * CD * CD);
  bf16_t* wb_eff1   = allocH((size_t)CL * CDFF * CD);
  bf16_t* wb_eff2   = allocH((size_t)CL * CD * CDFF);
  bf16_t* wb_saqkv  = allocH((size_t)CL * 3 * CD * CD);
  bf16_t* wb_saout  = allocH((size_t)CL * CD * CD);
  bf16_t* wb_caqkv  = allocH((size_t)CL * 3 * CD * CD);
  bf16_t* wb_caout  = allocH((size_t)CL * CD * CD);
  bf16_t* wb_dff1   = allocH((size_t)CL * CDFF * CD);
  bf16_t* wb_dff2   = allocH((size_t)CL * CD * CDFF);
  bf16_t* wb_cb     = allocH((size_t)CK * CD);
  bf16_t* wb_recW   = allocH((size_t)CS * CD);
  bf16_t* sb_states = allocH((size_t)CB * CT * CS);

  // ---- activations ----
  const size_t ME = (size_t)CB * CT;          // 2560 encoder rows
  const size_t MDMAX = (size_t)CB * (CT - 1); // 2432 decoder rows max
  float*  se_f   = allocF(ME * CD);
  float*  xf     = allocF(ME * CD);
  bf16_t* xb     = allocH(ME * CD);
  float*  qkvf   = allocF(ME * 3 * CD);
  float*  attnf  = allocF(ME * CD);
  bf16_t* attnb  = allocH(ME * CD);
  float*  tmpf   = allocF(ME * CD);
  float*  hf     = allocF(ME * CDFF);        // also reused as VQ dot buffer
  bf16_t* hb     = allocH(ME * CDFF);
  float*  quantf = allocF(ME * CD);
  float*  dec_in = allocF(ME * CD);          // [B,T,D]
  float*  yf     = allocF(MDMAX * CD);
  bf16_t* yb     = allocH(MDMAX * CD);
  bf16_t* memb   = allocH(MDMAX * CD);
  float*  cnorm  = allocF(CK);
  int*    idxb   = (int*)alloc((size_t)CB * CT * sizeof(int));
  float*  rowls  = allocF(ME);
  bf16_t* lastb  = allocH((size_t)CB * CD);
  float*  predf  = allocF((size_t)CB * CS);
  bf16_t* predb  = allocH((size_t)CB * CS);
  float*  bdtmp  = allocF((size_t)CB * CD);

  // ---- one-time (per call) fp32 -> bf16 weight conversions ----
  cvt(enc_W,     wb_encW,  (size_t)CD * CS);
  cvt(enc_qkv_w, wb_eqkv,  (size_t)CL * 3 * CD * CD);
  cvt(enc_out_w, wb_eout,  (size_t)CL * CD * CD);
  cvt(enc_ff1_w, wb_eff1,  (size_t)CL * CDFF * CD);
  cvt(enc_ff2_w, wb_eff2,  (size_t)CL * CD * CDFF);
  cvt(dsa_qkv_w, wb_saqkv, (size_t)CL * 3 * CD * CD);
  cvt(dsa_out_w, wb_saout, (size_t)CL * CD * CD);
  cvt(dca_qkv_w, wb_caqkv, (size_t)CL * 3 * CD * CD);
  cvt(dca_out_w, wb_caout, (size_t)CL * CD * CD);
  cvt(dff1_w,    wb_dff1,  (size_t)CL * CDFF * CD);
  cvt(dff2_w,    wb_dff2,  (size_t)CL * CD * CDFF);
  cvt(codebook,  wb_cb,    (size_t)CK * CD);
  cvt(rec_W,     wb_recW,  (size_t)CS * CD);
  cvt(states,    sb_states,(size_t)CB * CT * CS);

  // ================= Encoder =================
  gemm(sb_states, wb_encW, enc_b, se_f, nullptr, (int)ME, CD, CS, 0);
  embed_kernel<<<(CB * CT * CD + 255) / 256, 256, 0, stream>>>(
      se_f, actions, act_emb, pos_emb, xf, xb);

  for (int i = 0; i < CL; ++i) {
    gemm(xb, wb_eqkv + (size_t)i * 3 * CD * CD, enc_qkv_b + (size_t)i * 3 * CD,
         qkvf, nullptr, (int)ME, 3 * CD, CD, 0);
    attn(qkvf, 3 * CD, qkvf + CD, 3 * CD, qkvf + 2 * CD, 3 * CD, attnf, CT, CT, 0);
    cvt(attnf, attnb, ME * CD);
    gemm(attnb, wb_eout + (size_t)i * CD * CD, enc_out_b + (size_t)i * CD,
         tmpf, nullptr, (int)ME, CD, CD, 0);
    ln(xf, tmpf, enc_ln1_g + (size_t)i * CD, enc_ln1_b + (size_t)i * CD, xf, xb, (int)ME);
    gemm(xb, wb_eff1 + (size_t)i * CDFF * CD, enc_ff1_b + (size_t)i * CDFF,
         hf, hb, (int)ME, CDFF, CD, 1);
    gemm(hb, wb_eff2 + (size_t)i * CD * CDFF, enc_ff2_b + (size_t)i * CD,
         tmpf, nullptr, (int)ME, CD, CDFF, 0);
    ln(xf, tmpf, enc_ln2_g + (size_t)i * CD, enc_ln2_b + (size_t)i * CD, xf, xb, (int)ME);
  }

  // ================= Vector quantizer =================
  const size_t PREDN = (size_t)CB * (CT - 1) * CS;
  gemm(xb, wb_cb, nullptr, hf /*dots [ME,CK]*/, nullptr, (int)ME, CK, CD, 0);
  rownorm_kernel<<<CK, 256, 0, stream>>>(codebook, cnorm, CD);
  vq_argmin_kernel<<<(int)ME, 256, 0, stream>>>(hf, cnorm, idxb, CK);
  vq_quant_kernel<<<(int)ME, 256, 0, stream>>>(xf, codebook, idxb, quantf, rowls, CD);
  reduce_kernel<<<1, 256, 0, stream>>>(rowls, (int)ME,
      2.0f / ((float)CB * CT * CD), out + PREDN);          // (BETA+1)*mse
  idx_out_kernel<<<(CB * CT + 255) / 256, 256, 0, stream>>>(idxb, out + PREDN + 1, CB * CT);

  // ================= Autoregressive decoder =================
  for (int t = 0; t < CT - 1; ++t) {
    const int Tc = t + 1;
    const int M = CB * Tc;
    if (t == 0) {
      set_col_seq_kernel<<<(CB * CD + 255) / 256, 256, 0, stream>>>(se_f, dec_in, 0, 0);
    } else {
      gemm(predb, wb_encW, enc_b, bdtmp, nullptr, CB, CD, CS, 0);
      set_col_kernel<<<(CB * CD + 255) / 256, 256, 0, stream>>>(bdtmp, dec_in, t);
    }
    build_y_kernel<<<(M * CD + 255) / 256, 256, 0, stream>>>(dec_in, pos_emb, yf, yb, Tc);
    build_mem_kernel<<<(M * CD + 255) / 256, 256, 0, stream>>>(quantf, memb, Tc);

    for (int i = 0; i < CL; ++i) {
      // --- causal self-attention ---
      gemm(yb, wb_saqkv + (size_t)i * 3 * CD * CD, dsa_qkv_b + (size_t)i * 3 * CD,
           qkvf, nullptr, M, 3 * CD, CD, 0);
      attn(qkvf, 3 * CD, qkvf + CD, 3 * CD, qkvf + 2 * CD, 3 * CD, attnf, Tc, Tc, 1);
      cvt(attnf, attnb, (size_t)M * CD);
      gemm(attnb, wb_saout + (size_t)i * CD * CD, dsa_out_b + (size_t)i * CD,
           tmpf, nullptr, M, CD, CD, 0);
      ln(yf, tmpf, dln1_g + (size_t)i * CD, dln1_b + (size_t)i * CD, yf, yb, M);
      // --- cross-attention (q from y, k/v from quant memory) ---
      gemm(yb, wb_caqkv + (size_t)i * 3 * CD * CD, dca_qkv_b + (size_t)i * 3 * CD,
           qkvf, nullptr, M, CD, CD, 0);                         // wq rows 0..D-1
      gemm(memb, wb_caqkv + (size_t)i * 3 * CD * CD + (size_t)CD * CD,
           dca_qkv_b + (size_t)i * 3 * CD + CD,
           qkvf + (size_t)M * CD, nullptr, M, 2 * CD, CD, 0);    // wk;wv rows D..3D-1
      attn(qkvf, CD, qkvf + (size_t)M * CD, 2 * CD,
           qkvf + (size_t)M * CD + CD, 2 * CD, attnf, Tc, Tc, 0);
      cvt(attnf, attnb, (size_t)M * CD);
      gemm(attnb, wb_caout + (size_t)i * CD * CD, dca_out_b + (size_t)i * CD,
           tmpf, nullptr, M, CD, CD, 0);
      ln(yf, tmpf, dln2_g + (size_t)i * CD, dln2_b + (size_t)i * CD, yf, yb, M);
      // --- FFN ---
      gemm(yb, wb_dff1 + (size_t)i * CDFF * CD, dff1_b + (size_t)i * CDFF,
           hf, hb, M, CDFF, CD, 1);
      gemm(hb, wb_dff2 + (size_t)i * CD * CDFF, dff2_b + (size_t)i * CD,
           tmpf, nullptr, M, CD, CDFF, 0);
      ln(yf, tmpf, dln3_g + (size_t)i * CD, dln3_b + (size_t)i * CD, yf, yb, M);
    }

    gather_last_kernel<<<(CB * CD + 255) / 256, 256, 0, stream>>>(yf, lastb, Tc);
    gemm(lastb, wb_recW, rec_b, predf, nullptr, CB, CS, CD, 0);
    store_pred_kernel<<<(CB * CS + 255) / 256, 256, 0, stream>>>(predf, out, predb, t);
  }
}
